// MultiScaleRetention_74921409511439
// MI455X (gfx1250) — matrix operand order repaired and checked by reading, not compiled, not run on hardware
//
#include <hip/hip_runtime.h>
#include <hip/hip_bf16.h>
#include <math.h>

// ---------------------------------------------------------------------------
// MultiScaleRetention forward for MI455X (gfx1250), wave32 + WMMA bf16 + TDM.
// Pipeline: cvt -> proj(QKV) -> gate -> xpos -> fused retention -> out GEMM.
// ---------------------------------------------------------------------------

typedef __bf16 bf16_t;
typedef __attribute__((ext_vector_type(16))) __bf16 v16bf;
typedef __attribute__((ext_vector_type(8)))  __bf16 v8bf;
typedef __attribute__((ext_vector_type(8)))  float  v8f;
typedef __attribute__((ext_vector_type(4)))  unsigned int v4u;
typedef __attribute__((ext_vector_type(8)))  int v8i;
typedef __attribute__((ext_vector_type(4)))  int v4i;

#define L_SEQ   4096
#define D_MODEL 512
#define NHEADS  8
#define HD      64

union frag_u { struct { v8bf lo, hi; } p; v16bf v; };

__device__ __forceinline__ v16bf ld_frag(const bf16_t* p0, const bf16_t* p1) {
  frag_u u;
  u.p.lo = *(const v8bf*)p0;   // 8 contiguous bf16 (16B)
  u.p.hi = *(const v8bf*)p1;   // 8 contiguous bf16 (16B)
  return u.v;
}

__device__ __forceinline__ v8f wmma_bf16(v16bf a, v16bf b, v8f c) {
  // D = A(16x32) * B(32x16) + C, f32 accumulate
  return __builtin_amdgcn_wmma_f32_16x16x32_bf16(false, a, false, b, (short)0, c,
                                                 false, false);
}

// TDM: DMA one 32x64 bf16 tile (row stride 64 elements) from global into LDS.
// D# per CDNA5 ISA ch.8: group0 = {count=1, lds_addr, global_addr, type=2},
// group1 = {data_size=2B, tensor_dim0=64, tensor_dim1=32, tile 64x32,
// dim0_stride=64}; groups 2/3 zero (2-D tile). Tracked by TENSORcnt.
// This toolchain exposes the 6-arg builtin: (g0, g1, g2, g3, g4, cpol).
__device__ __forceinline__ void tdm_load_tile32x64(const bf16_t* gsrc,
                                                   unsigned lds_off) {
  unsigned long long ga = (unsigned long long)(uintptr_t)gsrc;
  v4u g0 = { 1u,                                   // count=1 (valid user D#)
             lds_off,                              // lds_addr (bytes)
             (unsigned)ga,                         // global_addr[31:0]
             ((unsigned)(ga >> 32) & 0x01ffffffu)  // global_addr[56:32]
               | 0x80000000u };                    // type=2 ("image")
  v8i g1 = { (int)0x00010000u,   // d0: wg_mask=0, data_size=1 (2 bytes)
             (int)(64u << 16),   // d1[31:16] = tensor_dim0 = 64
             (int)(32u << 16),   // d2[15:0]=dim0_hi=0, d2[31:16]=tensor_dim1=32
             (int)(64u << 16),   // d3[15:0]=dim1_hi=0, d3[31:16]=tile_dim0=64
             32,                 // d4[15:0]=tile_dim1=32, tile_dim2=0
             64,                 // d5 = tensor_dim0_stride = 64
             0, 0 };
  v4i gz4 = { 0, 0, 0, 0 };
  v8i gz8 = { 0, 0, 0, 0, 0, 0, 0, 0 };
  __builtin_amdgcn_tensor_load_to_lds(g0, g1, gz4, gz4, gz8, 0);
}

// ---------------------------------------------------------------------------
// 0) f32 -> bf16 conversion
// ---------------------------------------------------------------------------
__global__ void __launch_bounds__(256)
cvt_f32_bf16(const float* __restrict__ src, bf16_t* __restrict__ dst, int n) {
  int i = blockIdx.x * blockDim.x + threadIdx.x;
  if (i < n) dst[i] = (bf16_t)src[i];
}

// ---------------------------------------------------------------------------
// 1) QKV projection: per wave one 16x64 tile for one head, one of {q,k,v}.
//    q,k -> f32 scratch (rotary applied later); v -> bf16 directly.
// ---------------------------------------------------------------------------
__global__ void __launch_bounds__(256)
proj_kernel(const bf16_t* __restrict__ Xb,
            const bf16_t* __restrict__ Wqb,
            const bf16_t* __restrict__ Wkb,
            const bf16_t* __restrict__ Wvb,
            float* __restrict__ qf, float* __restrict__ kf,
            bf16_t* __restrict__ vb)
{
  const int lane = threadIdx.x & 31;
  const int wid  = threadIdx.x >> 5;
  const int mt   = blockIdx.x * 8 + wid;            // 16-row tile index, 0..255
  const int h    = blockIdx.y;
  const int mat  = blockIdx.z;                      // 0=q 1=k 2=v
  const int row  = mt * 16 + (lane & 15);
  const int off  = (lane >> 4) << 3;                // 0 or 8
  const int bsel = (lane & 15) + ((lane >> 4) << 4); // B-fragment row within 32

  const bf16_t* W = (mat == 0 ? Wqb : (mat == 1 ? Wkb : Wvb))
                    + (size_t)h * D_MODEL * HD;

  v8f zero8 = {0.f,0.f,0.f,0.f,0.f,0.f,0.f,0.f};
  v8f acc[4] = {zero8, zero8, zero8, zero8};

  for (int k0 = 0; k0 < D_MODEL; k0 += 32) {
    const bf16_t* xrow = Xb + (size_t)row * D_MODEL + k0;
    v16bf a = ld_frag(xrow + off, xrow + 16 + off);
    const bf16_t* wrow = W + (size_t)k0 * HD + (lane & 15);
#pragma unroll
    for (int t = 0; t < 4; ++t) {
      v16bf b;
#pragma unroll
      for (int s = 0; s < 16; ++s)
        b[s] = wrow[(size_t)(off + (s & 7) + ((s >> 3) << 4)) * HD + t * 16];
      acc[t] = wmma_bf16(a, b, acc[t]);
    }
  }

  const int rbase = mt * 16 + ((lane >> 4) << 3);
  const int ncol  = lane & 15;
  if (mat < 2) {
    float* dst = (mat == 0 ? qf : kf) + (size_t)h * L_SEQ * HD;
#pragma unroll
    for (int t = 0; t < 4; ++t)
#pragma unroll
      for (int r = 0; r < 8; ++r)
        dst[(size_t)(rbase + r) * HD + t * 16 + ncol] = acc[t][r];
  } else {
    bf16_t* dst = vb + (size_t)h * L_SEQ * HD;
#pragma unroll
    for (int t = 0; t < 4; ++t)
#pragma unroll
      for (int r = 0; r < 8; ++r)
        dst[(size_t)(rbase + r) * HD + t * 16 + ncol] = (bf16_t)acc[t][r];
  }
}

// ---------------------------------------------------------------------------
// 2) gate = silu(X @ W_G): per wave one 16x64 tile of the 4096x512 output.
// ---------------------------------------------------------------------------
__global__ void __launch_bounds__(256)
gate_kernel(const bf16_t* __restrict__ Xb, const bf16_t* __restrict__ Wgb,
            float* __restrict__ gate)
{
  const int lane = threadIdx.x & 31;
  const int wid  = threadIdx.x >> 5;
  const int mt   = blockIdx.x * 8 + wid;
  const int nb   = blockIdx.y * 64;
  const int row  = mt * 16 + (lane & 15);
  const int off  = (lane >> 4) << 3;
  const int bsel = (lane & 15) + ((lane >> 4) << 4);

  v8f zero8 = {0.f,0.f,0.f,0.f,0.f,0.f,0.f,0.f};
  v8f acc[4] = {zero8, zero8, zero8, zero8};

  for (int k0 = 0; k0 < D_MODEL; k0 += 32) {
    const bf16_t* xrow = Xb + (size_t)row * D_MODEL + k0;
    v16bf a = ld_frag(xrow + off, xrow + 16 + off);
    const bf16_t* wrow = Wgb + (size_t)k0 * D_MODEL + nb + (lane & 15);
#pragma unroll
    for (int t = 0; t < 4; ++t) {
      v16bf b;
#pragma unroll
      for (int s = 0; s < 16; ++s)
        b[s] = wrow[(size_t)(off + (s & 7) + ((s >> 3) << 4)) * D_MODEL + t * 16];
      acc[t] = wmma_bf16(a, b, acc[t]);
    }
  }

  const int rbase = mt * 16 + ((lane >> 4) << 3);
  const int ncol  = lane & 15;
#pragma unroll
  for (int t = 0; t < 4; ++t)
#pragma unroll
    for (int r = 0; r < 8; ++r) {
      float x = acc[t][r];
      float s = x / (1.f + __expf(-x));              // silu
      gate[(size_t)(rbase + r) * D_MODEL + nb + t * 16 + ncol] = s;
    }
}

// ---------------------------------------------------------------------------
// 3) xPos rotary on q (scale) and k (1/scale); k stored transposed [h][d][L]
//    so the S=QK^T B-fragment is contiguous; q stored row-major [h][L][64].
// ---------------------------------------------------------------------------
__global__ void __launch_bounds__(256)
xpos_kernel(const float* __restrict__ qf, const float* __restrict__ kf,
            bf16_t* __restrict__ qb, bf16_t* __restrict__ kTb)
{
  int idx = blockIdx.x * blockDim.x + threadIdx.x;  // H * L * 32 threads
  int j   = idx & 31;                               // rotary pair index
  int pos = (idx >> 5) & (L_SEQ - 1);
  int h   = idx >> 17;

  float sv    = (2.f * j + 0.4f * HD) / (1.4f * HD);
  float scale = exp2f(log2f(sv) * ((float)pos * (1.f / 512.f)));
  float invf  = exp2f(log2f(10000.f) * (-(float)j / 32.f));
  float sinu  = (float)pos * invf;
  float s, c;
  sincosf(sinu, &s, &c);

  size_t base = ((size_t)h * L_SEQ + pos) * HD + 2 * j;
  float q1 = qf[base], q2 = qf[base + 1];
  float cs = c * scale, ss = s * scale;
  qb[base]     = (bf16_t)(q1 * cs - q2 * ss);
  qb[base + 1] = (bf16_t)(q2 * cs + q1 * ss);

  float k1 = kf[base], k2 = kf[base + 1];
  float inv = 1.f / scale;
  float ck = c * inv, sk = s * inv;
  size_t tb = ((size_t)h * HD + 2 * j) * (size_t)L_SEQ + pos;
  kTb[tb]         = (bf16_t)(k1 * ck - k2 * sk);
  kTb[tb + L_SEQ] = (bf16_t)(k2 * ck + k1 * sk);
}

// ---------------------------------------------------------------------------
// 4) Fused retention core + groupnorm + gate. One wave = one 16-row Q tile of
//    one head; iterates causal 32-key chunks. Scores never touch HBM.
//    V chunks are double-buffered into LDS by the Tensor Data Mover, issued
//    one chunk ahead and synchronized with s_wait_tensorcnt.
// ---------------------------------------------------------------------------
__global__ void __launch_bounds__(256)
retention_kernel(const bf16_t* __restrict__ qb, const bf16_t* __restrict__ kTb,
                 const bf16_t* __restrict__ vb, const float* __restrict__ gate,
                 const float* __restrict__ gnw, const float* __restrict__ gnb,
                 bf16_t* __restrict__ Yb)
{
  __shared__ bf16_t sstage[8][16][40] __attribute__((aligned(16)));
  __shared__ bf16_t vstage[8][2][32][64] __attribute__((aligned(16)));

  const int lane = threadIdx.x & 31;
  const int wid  = threadIdx.x >> 5;
  const int mt   = blockIdx.x * 8 + wid;            // 0..255
  const int h    = blockIdx.y;
  const int q0   = mt * 16;
  const int off  = (lane >> 4) << 3;
  const int dsel = (lane & 15) + ((lane >> 4) << 4);
  const int ncol = lane & 15;
  const int half8 = (lane >> 4) << 3;

  // gamma_h = 1 - exp(linspace(log(1/32), log(1/512), 8))[h]
  float lgam   = -3.4657359f + (float)h * (-0.39608410f);
  float gamma  = 1.f - __expf(lgam);
  float l2g    = log2f(gamma);

  // Q fragments for both 32-wide d-steps (row-major, stride 64 halves)
  const bf16_t* qrow = qb + ((size_t)h * L_SEQ + (q0 + (lane & 15))) * HD;
  v16bf aq0 = ld_frag(qrow + off,      qrow + 16 + off);
  v16bf aq1 = ld_frag(qrow + 32 + off, qrow + 48 + off);

  const bf16_t* kTh = kTb + (size_t)h * HD * L_SEQ;
  const bf16_t* vh  = vb  + (size_t)h * L_SEQ * HD;

  v8f zero8 = {0.f,0.f,0.f,0.f,0.f,0.f,0.f,0.f};
  v8f acc[4] = {zero8, zero8, zero8, zero8};

  const int mlast = ((q0 + 15) >> 5) << 5;

  // TDM prologue: stage V chunk 0 into LDS buffer 0
  tdm_load_tile32x64(vh, (unsigned)(uintptr_t)(void*)&vstage[wid][0][0][0]);

  for (int m0 = 0; m0 <= mlast; m0 += 32) {
    const int  cur  = (m0 >> 5) & 1;
    const bool more = (m0 + 32) <= mlast;
    if (more)  // async-prefetch next V chunk into the other buffer
      tdm_load_tile32x64(vh + (size_t)(m0 + 32) * HD,
                         (unsigned)(uintptr_t)(void*)&vstage[wid][cur ^ 1][0][0]);

    // ---- S = Q K^T for keys [m0, m0+32) : 4 WMMAs
    const bf16_t* k0p = kTh + m0 + ncol;                         // d-step 0
    const bf16_t* k1p = kTh + (size_t)32 * L_SEQ + m0 + ncol;    // d-step 1
    __builtin_prefetch(k0p + 32, 0, 3);
    __builtin_prefetch(k1p + 32, 0, 3);

    v8f s0 = zero8, s1 = zero8;
    v16bf b00, b01, b10, b11;
#pragma unroll
    for (int s = 0; s < 16; ++s) {
      size_t kk = (size_t)(off + (s & 7) + ((s >> 3) << 4)) * L_SEQ;
      b00[s] = k0p[kk];      b01[s] = k0p[kk + 16];
      b10[s] = k1p[kk];      b11[s] = k1p[kk + 16];
    }
    s0 = wmma_bf16(aq0, b00, s0);  s0 = wmma_bf16(aq1, b10, s0);
    s1 = wmma_bf16(aq0, b01, s1);  s1 = wmma_bf16(aq1, b11, s1);

    // ---- causal decay mask in-register, stage to LDS in A-matrix layout
#pragma unroll
    for (int r = 0; r < 8; ++r) {
      int n  = q0 + half8 + r;
      int m  = m0 + ncol;
      int m2 = m + 16;
      float d0 = (n >= m)  ? exp2f(l2g * (float)(n - m))  : 0.f;
      float d1 = (n >= m2) ? exp2f(l2g * (float)(n - m2)) : 0.f;
      sstage[wid][half8 + r][ncol]      = (bf16_t)(s0[r] * d0);
      sstage[wid][half8 + r][ncol + 16] = (bf16_t)(s1[r] * d1);
    }
    asm volatile("s_wait_dscnt 0" ::: "memory");   // intra-wave LDS RAW

    const bf16_t* srow = &sstage[wid][lane & 15][0];
    v16bf as = ld_frag(srow + off, srow + 16 + off);

    // ---- wait for current V tile (TDM in-order per wave: cnt<=1 => landed)
    if (more) __builtin_amdgcn_s_wait_tensorcnt(1);
    else      __builtin_amdgcn_s_wait_tensorcnt(0);
    asm volatile("" ::: "memory");

    // ---- acc += S_decayed * V : 4 WMMAs (V B-fragments from LDS)
    const bf16_t* vrow = &vstage[wid][cur][0][ncol];
#pragma unroll
    for (int t = 0; t < 4; ++t) {
      v16bf bv;
#pragma unroll
      for (int s = 0; s < 16; ++s)
        bv[s] = vrow[(off + (s & 7) + ((s >> 3) << 4)) * 64 + t * 16];
      acc[t] = wmma_bf16(as, bv, acc[t]);
    }
  }

  // ---- groupnorm over the 64 v-dims of each row (rows live per 16-lane half)
  float sum[8], sq[8];
#pragma unroll
  for (int r = 0; r < 8; ++r) {
    float s = 0.f, q = 0.f;
#pragma unroll
    for (int t = 0; t < 4; ++t) { float x = acc[t][r]; s += x; q += x * x; }
    sum[r] = s; sq[r] = q;
  }
#pragma unroll
  for (int mask = 1; mask < 16; mask <<= 1) {
#pragma unroll
    for (int r = 0; r < 8; ++r) {
      sum[r] += __shfl_xor(sum[r], mask, 32);
      sq[r]  += __shfl_xor(sq[r],  mask, 32);
    }
  }

#pragma unroll
  for (int r = 0; r < 8; ++r) {
    float mu   = sum[r] * (1.f / 64.f);
    float var  = sq[r] * (1.f / 64.f) - mu * mu;
    float rstd = rsqrtf(var + 1e-5f);
    int rr = q0 + half8 + r;
#pragma unroll
    for (int t = 0; t < 4; ++t) {
      int col = h * HD + t * 16 + ncol;
      float g  = (acc[t][r] - mu) * rstd * gnw[col] + gnb[col];
      float gv = gate[(size_t)rr * D_MODEL + col];
      Yb[(size_t)rr * D_MODEL + col] = (bf16_t)(g * gv);
    }
  }
}

// ---------------------------------------------------------------------------
// 5) out = Y @ W_O  (4096x512 x 512x512), f32 output
// ---------------------------------------------------------------------------
__global__ void __launch_bounds__(256)
out_kernel(const bf16_t* __restrict__ Yb, const bf16_t* __restrict__ Wob,
           float* __restrict__ out)
{
  const int lane = threadIdx.x & 31;
  const int wid  = threadIdx.x >> 5;
  const int mt   = blockIdx.x * 8 + wid;
  const int nb   = blockIdx.y * 64;
  const int row  = mt * 16 + (lane & 15);
  const int off  = (lane >> 4) << 3;
  const int bsel = (lane & 15) + ((lane >> 4) << 4);

  v8f zero8 = {0.f,0.f,0.f,0.f,0.f,0.f,0.f,0.f};
  v8f acc[4] = {zero8, zero8, zero8, zero8};

  for (int k0 = 0; k0 < D_MODEL; k0 += 32) {
    const bf16_t* yrow = Yb + (size_t)row * D_MODEL + k0;
    v16bf a = ld_frag(yrow + off, yrow + 16 + off);
    const bf16_t* wrow = Wob + (size_t)k0 * D_MODEL + nb + (lane & 15);
#pragma unroll
    for (int t = 0; t < 4; ++t) {
      v16bf b;
#pragma unroll
      for (int s = 0; s < 16; ++s)
        b[s] = wrow[(size_t)(off + (s & 7) + ((s >> 3) << 4)) * D_MODEL + t * 16];
      acc[t] = wmma_bf16(a, b, acc[t]);
    }
  }

  const int rbase = mt * 16 + ((lane >> 4) << 3);
  const int ncol  = lane & 15;
#pragma unroll
  for (int t = 0; t < 4; ++t)
#pragma unroll
    for (int r = 0; r < 8; ++r)
      out[(size_t)(rbase + r) * D_MODEL + nb + t * 16 + ncol] = acc[t][r];
}

// ---------------------------------------------------------------------------
// Host launcher
// ---------------------------------------------------------------------------
extern "C" void kernel_launch(void* const* d_in, const int* in_sizes, int n_in,
                              void* d_out, int out_size, void* d_ws, size_t ws_size,
                              hipStream_t stream)
{
  const float* X   = (const float*)d_in[0];
  const float* W_Q = (const float*)d_in[1];
  const float* W_K = (const float*)d_in[2];
  const float* W_V = (const float*)d_in[3];
  const float* W_G = (const float*)d_in[4];
  const float* W_O = (const float*)d_in[5];
  const float* gnw = (const float*)d_in[6];
  const float* gnb = (const float*)d_in[7];
  float* out = (float*)d_out;

  constexpr size_t NX   = (size_t)L_SEQ * D_MODEL;        // 2,097,152
  constexpr size_t NWH  = (size_t)NHEADS * D_MODEL * HD;  //   262,144
  constexpr size_t NWG  = (size_t)D_MODEL * D_MODEL;      //   262,144
  constexpr size_t NQKV = (size_t)NHEADS * L_SEQ * HD;    // 2,097,152

  char* ws = (char*)d_ws;
  size_t off = 0;
  auto take = [&](size_t bytes) {
    size_t r = off; off += (bytes + 255) & ~(size_t)255; return r;
  };
  bf16_t* Xb   = (bf16_t*)(ws + take(NX  * 2));
  bf16_t* Wqb  = (bf16_t*)(ws + take(NWH * 2));
  bf16_t* Wkb  = (bf16_t*)(ws + take(NWH * 2));
  bf16_t* Wvb  = (bf16_t*)(ws + take(NWH * 2));
  bf16_t* Wgb  = (bf16_t*)(ws + take(NWG * 2));
  bf16_t* Wob  = (bf16_t*)(ws + take(NWG * 2));
  float*  qfp  = (float*)(ws + take(NQKV * 4));
  float*  kfp  = (float*)(ws + take(NQKV * 4));
  float*  gatb = (float*)(ws + take(NX  * 4));
  bf16_t* qbuf = (bf16_t*)(ws + take(NQKV * 2));
  bf16_t* kTb  = (bf16_t*)(ws + take(NQKV * 2));
  bf16_t* vbuf = (bf16_t*)(ws + take(NQKV * 2));
  bf16_t* Yb   = (bf16_t*)(ws + take(NX  * 2));
  (void)ws_size; (void)in_sizes; (void)n_in; (void)out_size;

  // 0) bf16 conversions
  cvt_f32_bf16<<<(int)((NX  + 255) / 256), 256, 0, stream>>>(X,   Xb,  (int)NX);
  cvt_f32_bf16<<<(int)((NWH + 255) / 256), 256, 0, stream>>>(W_Q, Wqb, (int)NWH);
  cvt_f32_bf16<<<(int)((NWH + 255) / 256), 256, 0, stream>>>(W_K, Wkb, (int)NWH);
  cvt_f32_bf16<<<(int)((NWH + 255) / 256), 256, 0, stream>>>(W_V, Wvb, (int)NWH);
  cvt_f32_bf16<<<(int)((NWG + 255) / 256), 256, 0, stream>>>(W_G, Wgb, (int)NWG);
  cvt_f32_bf16<<<(int)((NWG + 255) / 256), 256, 0, stream>>>(W_O, Wob, (int)NWG);

  // 1) QKV projections (WMMA)
  proj_kernel<<<dim3(32, NHEADS, 3), 256, 0, stream>>>(Xb, Wqb, Wkb, Wvb,
                                                       qfp, kfp, vbuf);
  // 2) gate = silu(X @ W_G) (WMMA)
  gate_kernel<<<dim3(32, 8), 256, 0, stream>>>(Xb, Wgb, gatb);

  // 3) xPos rotary + K transpose (H*L*32 threads)
  xpos_kernel<<<4096, 256, 0, stream>>>(qfp, kfp, qbuf, kTb);

  // 4) fused retention + decay + groupnorm + gate (WMMA + TDM double-buffer)
  retention_kernel<<<dim3(32, NHEADS), 256, 0, stream>>>(qbuf, kTb, vbuf, gatb,
                                                         gnw, gnb, Yb);
  // 5) output GEMM (WMMA)
  out_kernel<<<dim3(32, 8), 256, 0, stream>>>(Yb, Wob, out);
}
